// SimplifiedTimeEquivConvLayer_68487548502116
// MI455X (gfx1250) — matrix operand order repaired and checked by reading, not compiled, not run on hardware
//
#include <hip/hip_runtime.h>
#include <hip/hip_bf16.h>

#define B_DIM 4
#define N_NODE 10000
#define E_EDGE 100000
#define C_IN 16
#define NEF 128
#define MUL0 16
#define MUL1 8
#define W_NUMEL 384          // 256 + 128
#define OUT_DIM 40           // 16 + 24
#define ALPHA 0.25f          // 1/sqrt(16)
#define BN_EPS 1e-5f
#define TILE_E 32            // edges per block (two 16-row WMMA M-tiles)

typedef __attribute__((ext_vector_type(16))) __bf16 v16bf;
typedef __attribute__((ext_vector_type(8)))  float  v8f;
typedef __attribute__((ext_vector_type(8)))  unsigned short v8us;

union BfVec { v16bf bf; v8us u8[2]; };

__device__ __forceinline__ unsigned short f2bf(float f) {
    union { float f; unsigned u; } v; v.f = f;
    unsigned u = v.u;
    unsigned r = u + 0x7FFFu + ((u >> 16) & 1u);   // round-to-nearest-even
    return (unsigned short)(r >> 16);
}

// ---------------- workspace layout (bytes) ----------------
static constexpr size_t OFF_XS   = 0;                          // 4*10000*16 f32 = 2,560,000
static constexpr size_t OFF_W1T  = OFF_XS  + 2560000;          // 128*128 bf16   = 32,768
static constexpr size_t OFF_W2T  = OFF_W1T + 32768;            // 384*128 bf16   = 98,304
static constexpr size_t OFF_SEG  = OFF_W2T + 98304;            // 4*10000*40 f32 = 6,400,000
static constexpr size_t OFF_CNT  = OFF_SEG + 6400000;          // 10000 f32      = 40,000
static constexpr size_t OFF_PART = OFF_CNT + 40000;            // 80 f32
// seg + cnt + part are contiguous -> zeroed by one kernel

// ---------------- dynamic LDS layout for k_main (bytes) ----------------
static constexpr unsigned L_EW  = 0;                      // 32*384 f32 = 49152
static constexpr unsigned L_A   = L_EW + 32 * 384 * 4;    // 32*128 bf16 = 8192
static constexpr unsigned L_H   = L_A  + 32 * 128 * 2;    // 32*128 bf16 = 8192
static constexpr unsigned L_XG  = L_H  + 32 * 128 * 2;    // 32*16 f32   = 2048
static constexpr unsigned L_SH  = L_XG + 32 * 16 * 4;     // 32*4 f32    = 512
static constexpr unsigned L_DST = L_SH + 32 * 4 * 4;      // 32 int      = 128
static constexpr unsigned L_TOT = L_DST + 32 * 4;         // 68224 B (< 320KB/WGP)

// ---------------- tiny kernels ----------------
__global__ void k_zero(float* p, int n) {
    int i = blockIdx.x * blockDim.x + threadIdx.x;
    if (i < n) p[i] = 0.f;
}

// transpose+convert weights to bf16 [N][K]
__global__ void k_wcvt(const float* __restrict__ w1, const float* __restrict__ w2,
                       unsigned short* __restrict__ w1t, unsigned short* __restrict__ w2t) {
    int idx = blockIdx.x * blockDim.x + threadIdx.x;
    if (idx < NEF * NEF) {
        int k = idx >> 7, n = idx & 127;                 // w1[k][n]
        w1t[n * NEF + k] = f2bf(w1[idx]);
    } else if (idx < NEF * NEF + NEF * W_NUMEL) {
        int j = idx - NEF * NEF;
        int k = j / W_NUMEL, n = j - k * W_NUMEL;        // w2[k][n]
        w2t[(size_t)n * NEF + k] = f2bf(w2[j]);
    }
}

// closed-form 4-point rfft/irfft spectral mix
__global__ void k_fft(const float* __restrict__ x, const float* __restrict__ wr,
                      const float* __restrict__ wi, float* __restrict__ xs) {
    int idx = blockIdx.x * blockDim.x + threadIdx.x;
    if (idx >= N_NODE * MUL0) return;
    int n = idx >> 4, o = idx & 15;
    float reF0 = 0.f, reF1 = 0.f, imF1 = 0.f;
#pragma unroll
    for (int i = 0; i < C_IN; ++i) {
        float x0 = x[((size_t)0 * N_NODE + n) * C_IN + i];
        float x1 = x[((size_t)1 * N_NODE + n) * C_IN + i];
        float x2 = x[((size_t)2 * N_NODE + n) * C_IN + i];
        float x3 = x[((size_t)3 * N_NODE + n) * C_IN + i];
        float s = x0 + x1 + x2 + x3, u = x0 - x2, v = x1 - x3;
        float wr0 = wr[i * 32 + o * 2 + 0];
        float wr1 = wr[i * 32 + o * 2 + 1];
        float wi1 = wi[i * 32 + o * 2 + 1];
        reF0 += s * wr0;
        reF1 += u * wr1 + v * wi1;
        imF1 += u * wi1 - v * wr1;
    }
    xs[((size_t)0 * N_NODE + n) * MUL0 + o] = 0.25f * (reF0 + 2.f * reF1);
    xs[((size_t)1 * N_NODE + n) * MUL0 + o] = 0.25f * (reF0 - 2.f * imF1);
    xs[((size_t)2 * N_NODE + n) * MUL0 + o] = 0.25f * (reF0 - 2.f * reF1);
    xs[((size_t)3 * N_NODE + n) * MUL0 + o] = 0.25f * (reF0 + 2.f * imF1);
}

__global__ void k_cnt(const int* __restrict__ ei, float* __restrict__ cnt) {
    int e = blockIdx.x * blockDim.x + threadIdx.x;
    if (e < E_EDGE) atomicAdd(&cnt[ei[E_EDGE + e]], 1.f);
}

// ---------------- fused edge MLP (WMMA) + message + scatter ----------------
// 32 edges per block: 2 M-tiles share every weight B-fragment.
__global__ void __launch_bounds__(128)
k_main(const float* __restrict__ eattr, const float* __restrict__ esh,
       const int* __restrict__ ei, const float* __restrict__ xs,
       const unsigned short* __restrict__ w1t, const unsigned short* __restrict__ w2t,
       const float* __restrict__ b1, const float* __restrict__ b2,
       float* __restrict__ seg) {
    extern __shared__ char smem[];
    float*          sEW = (float*)(smem + L_EW);            // [32][384]
    unsigned short* sA  = (unsigned short*)(smem + L_A);    // [32][128] bf16
    unsigned short* sH  = (unsigned short*)(smem + L_H);    // [32][128] bf16
    float*          sXG = (float*)(smem + L_XG);            // [32][16]
    float*          sSH = (float*)(smem + L_SH);            // [32][4]
    int*            sDST= (int*)(smem + L_DST);             // [32]

    const int tid  = threadIdx.x;
    const int lane = tid & 31;
    const int wid  = tid >> 5;
    const int ksel = lane >> 4;          // 0 or 1 (half-wave K split)
    const int mrow = lane & 15;
    const int rowbase = ksel * 8;
    const int b    = blockIdx.y;
    const int e0   = blockIdx.x * TILE_E;

    // --- stage 0: loads (attr->bf16 LDS, gathered x, sh, dst) ---
    for (int i = tid; i < TILE_E * 128; i += 128) {
        int e = i >> 7, k = i & 127;
        sA[i] = f2bf(eattr[(((size_t)b * E_EDGE) + e0 + e) * NEF + k]);
    }
    for (int i = tid; i < TILE_E * 16; i += 128) {
        int e = i >> 4, c = i & 15;
        int srcv = ei[e0 + e];
        sXG[i] = xs[((size_t)b * N_NODE + srcv) * MUL0 + c];
    }
    if (tid < TILE_E * 4) {
        int e = tid >> 2, m = tid & 3;
        sSH[tid] = esh[(((size_t)b * E_EDGE) + e0 + e) * 4 + m];
    }
    if (tid < TILE_E) sDST[tid] = ei[E_EDGE + e0 + tid];
    __syncthreads();

    // --- GEMM1: h = relu(attr @ w1 + b1), 32x128, K=128 ---
    {
        int col0 = wid * 32 + mrow;
        int col1 = col0 + 16;
        v8f a00 = {}, a01 = {}, a10 = {}, a11 = {};
        for (int kk = 0; kk < 128; kk += 32) {
            BfVec A0, A1;
            const unsigned short* ar0 = &sA[mrow * 128 + kk + ksel * 8];
            const unsigned short* ar1 = ar0 + 16 * 128;
            A0.u8[0] = *(const v8us*)(ar0); A0.u8[1] = *(const v8us*)(ar0 + 16);
            A1.u8[0] = *(const v8us*)(ar1); A1.u8[1] = *(const v8us*)(ar1 + 16);
            BfVec B0, B1;
            const unsigned short* bp0 = w1t + (size_t)col0 * NEF + kk + ksel * 8;
            const unsigned short* bp1 = w1t + (size_t)col1 * NEF + kk + ksel * 8;
            B0.u8[0] = *(const v8us*)(bp0); B0.u8[1] = *(const v8us*)(bp0 + 16);
            B1.u8[0] = *(const v8us*)(bp1); B1.u8[1] = *(const v8us*)(bp1 + 16);
            a00 = __builtin_amdgcn_wmma_f32_16x16x32_bf16(false, A0.bf, false, B0.bf,
                                                          (short)0, a00, false, false);
            a01 = __builtin_amdgcn_wmma_f32_16x16x32_bf16(false, A0.bf, false, B1.bf,
                                                          (short)0, a01, false, false);
            a10 = __builtin_amdgcn_wmma_f32_16x16x32_bf16(false, A1.bf, false, B0.bf,
                                                          (short)0, a10, false, false);
            a11 = __builtin_amdgcn_wmma_f32_16x16x32_bf16(false, A1.bf, false, B1.bf,
                                                          (short)0, a11, false, false);
        }
        float bia0 = b1[col0], bia1 = b1[col1];
#pragma unroll
        for (int j = 0; j < 8; ++j) {
            int r = rowbase + j;
            sH[r * 128 + col0]         = f2bf(fmaxf(a00[j] + bia0, 0.f));
            sH[r * 128 + col1]         = f2bf(fmaxf(a01[j] + bia1, 0.f));
            sH[(16 + r) * 128 + col0]  = f2bf(fmaxf(a10[j] + bia0, 0.f));
            sH[(16 + r) * 128 + col1]  = f2bf(fmaxf(a11[j] + bia1, 0.f));
        }
    }
    __syncthreads();

    // --- GEMM2: ew = h @ w2 + b2, 32x384, K=128 ---
    for (int t = 0; t < 6; ++t) {
        int colg = (wid * 6 + t) * 16 + mrow;
        v8f c0 = {}, c1 = {};
        for (int kk = 0; kk < 128; kk += 32) {
            BfVec A0, A1;
            const unsigned short* ar0 = &sH[mrow * 128 + kk + ksel * 8];
            const unsigned short* ar1 = ar0 + 16 * 128;
            A0.u8[0] = *(const v8us*)(ar0); A0.u8[1] = *(const v8us*)(ar0 + 16);
            A1.u8[0] = *(const v8us*)(ar1); A1.u8[1] = *(const v8us*)(ar1 + 16);
            BfVec Bm;
            const unsigned short* bp = w2t + (size_t)colg * NEF + kk + ksel * 8;
            Bm.u8[0] = *(const v8us*)(bp); Bm.u8[1] = *(const v8us*)(bp + 16);
            c0 = __builtin_amdgcn_wmma_f32_16x16x32_bf16(false, A0.bf, false, Bm.bf,
                                                         (short)0, c0, false, false);
            c1 = __builtin_amdgcn_wmma_f32_16x16x32_bf16(false, A1.bf, false, Bm.bf,
                                                         (short)0, c1, false, false);
        }
        float bia = b2[colg];
#pragma unroll
        for (int j = 0; j < 8; ++j) {
            int r = rowbase + j;
            sEW[r * 384 + colg]        = c0[j] + bia;
            sEW[(16 + r) * 384 + colg] = c1[j] + bia;
        }
    }
    __syncthreads();

    // --- message + scatter-add (mean divide happens later) ---
    for (int idx = tid; idx < TILE_E * OUT_DIM; idx += 128) {
        int e = idx / OUT_DIM, c = idx - e * OUT_DIM;
        const float* xgr = &sXG[e * 16];
        const float* ewr = &sEW[e * 384];
        float v;
        if (c < MUL0) {
            float s = 0.f;
#pragma unroll
            for (int i = 0; i < C_IN; ++i) s += xgr[i] * ewr[i * MUL0 + c];
            v = ALPHA * sSH[e * 4 + 0] * s;
        } else {
            int q = c - MUL0, o = q / 3, m = q - 3 * o;
            float s = 0.f;
#pragma unroll
            for (int i = 0; i < C_IN; ++i) s += xgr[i] * ewr[256 + i * MUL1 + o];
            v = ALPHA * sSH[e * 4 + 1 + m] * s;
        }
        atomicAdd(&seg[((size_t)b * N_NODE + sDST[e]) * OUT_DIM + c], v);
    }
}

// ---------------- epilogue ----------------
__global__ void k_div(float* __restrict__ seg, const float* __restrict__ cnt) {
    int idx = blockIdx.x * blockDim.x + threadIdx.x;
    if (idx >= B_DIM * N_NODE * OUT_DIM) return;
    int n = (idx / OUT_DIM) % N_NODE;
    seg[idx] *= 1.f / fmaxf(cnt[n], 1.f);
}

__global__ void k_bnred(const float* __restrict__ agg, float* __restrict__ part) {
    int c = blockIdx.x, tid = threadIdx.x;
    float s = 0.f, s2 = 0.f;
    for (int r = tid; r < B_DIM * N_NODE; r += 256) {
        float v = agg[(size_t)r * OUT_DIM + c];
        s += v; s2 += v * v;
    }
    __shared__ float bs[256], bs2[256];
    bs[tid] = s; bs2[tid] = s2; __syncthreads();
    for (int st = 128; st > 0; st >>= 1) {
        if (tid < st) { bs[tid] += bs[tid + st]; bs2[tid] += bs2[tid + st]; }
        __syncthreads();
    }
    if (tid == 0) { part[c] = bs[0]; part[40 + c] = bs2[0]; }
}

__global__ void k_norm(const float* __restrict__ agg, const float* __restrict__ part,
                       const float* __restrict__ gamma, const float* __restrict__ beta,
                       float* __restrict__ out) {
    int idx = blockIdx.x * blockDim.x + threadIdx.x;
    if (idx >= B_DIM * N_NODE * OUT_DIM) return;
    int c = idx % OUT_DIM;
    const float inv = 1.f / (float)(B_DIM * N_NODE);
    float mu = part[c] * inv;
    float var = part[40 + c] * inv - mu * mu;
    out[idx] = (agg[idx] - mu) * rsqrtf(var + BN_EPS) * gamma[c] + beta[c];
}

extern "C" void kernel_launch(void* const* d_in, const int* in_sizes, int n_in,
                              void* d_out, int out_size, void* d_ws, size_t ws_size,
                              hipStream_t stream) {
    (void)in_sizes; (void)n_in; (void)out_size; (void)ws_size;
    const float* x     = (const float*)d_in[0];
    const int*   ei    = (const int*)  d_in[1];
    const float* eattr = (const float*)d_in[2];
    const float* esh   = (const float*)d_in[3];
    const float* wr    = (const float*)d_in[4];
    const float* wi    = (const float*)d_in[5];
    const float* w1    = (const float*)d_in[6];
    const float* b1    = (const float*)d_in[7];
    const float* w2    = (const float*)d_in[8];
    const float* b2    = (const float*)d_in[9];
    const float* gamma = (const float*)d_in[10];
    const float* beta  = (const float*)d_in[11];
    float* out = (float*)d_out;

    char* ws = (char*)d_ws;
    float*          xs  = (float*)(ws + OFF_XS);
    unsigned short* w1t = (unsigned short*)(ws + OFF_W1T);
    unsigned short* w2t = (unsigned short*)(ws + OFF_W2T);
    float*          seg = (float*)(ws + OFF_SEG);
    float*          cnt = (float*)(ws + OFF_CNT);
    float*          part= (float*)(ws + OFF_PART);

    const int zn = B_DIM * N_NODE * OUT_DIM + N_NODE + 80;   // seg+cnt+part contiguous
    k_zero<<<(zn + 255) / 256, 256, 0, stream>>>(seg, zn);

    const int wn = NEF * NEF + NEF * W_NUMEL;
    k_wcvt<<<(wn + 255) / 256, 256, 0, stream>>>(w1, w2, w1t, w2t);

    k_fft<<<(N_NODE * MUL0 + 255) / 256, 256, 0, stream>>>(x, wr, wi, xs);
    k_cnt<<<(E_EDGE + 255) / 256, 256, 0, stream>>>(ei, cnt);

    k_main<<<dim3(E_EDGE / TILE_E, B_DIM), 128, L_TOT, stream>>>(eattr, esh, ei, xs,
                                                                 w1t, w2t, b1, b2, seg);

    const int tn = B_DIM * N_NODE * OUT_DIM;
    k_div<<<(tn + 255) / 256, 256, 0, stream>>>(seg, cnt);
    k_bnred<<<OUT_DIM, 256, 0, stream>>>(seg, part);
    k_norm<<<(tn + 255) / 256, 256, 0, stream>>>(seg, part, gamma, beta, out);
}
